// GraphAttention_66503273611948
// MI455X (gfx1250) — matrix operand order repaired and checked by reading, not compiled, not run on hardware
//
#include <hip/hip_runtime.h>
#include <hip/hip_bf16.h>

typedef __attribute__((ext_vector_type(16))) __bf16 v16bf;
typedef __attribute__((ext_vector_type(8)))  float  v8f;

#define NN 4096
#define NF 128
#define TS 40   // LDS tile row stride in bf16 (80 B: 16B-aligned, conflict-free)

// ---------- helpers ----------
static __device__ __forceinline__ __bf16 f2bf(float f) {
  unsigned u = __builtin_bit_cast(unsigned, f);
  unsigned r = u + 0x7FFFu + ((u >> 16) & 1u);   // round-to-nearest-even
  unsigned short h = (unsigned short)(r >> 16);
  return __builtin_bit_cast(__bf16, h);
}

// ---------- elementwise kernels ----------
__global__ void ga_cvt_bf16(const float* __restrict__ in, __bf16* __restrict__ out, int n) {
  int i = blockIdx.x * blockDim.x + threadIdx.x;
  if (i < n) out[i] = f2bf(in[i]);
}

// transpose-convert: in [R x C] fp32 -> out [C x R] bf16
__global__ void ga_cvtT_bf16(const float* __restrict__ in, __bf16* __restrict__ out,
                             int R, int C) {
  int i = blockIdx.x * blockDim.x + threadIdx.x;
  if (i < R * C) {
    int r = i / C, c = i % C;
    out[(size_t)c * R + r] = f2bf(in[i]);
  }
}

// s_col[r] = dot(XW[r,:], a[0:128]);  s_row[r] = dot(XW[r,:], a[128:256])
__global__ void ga_row_scores(const float* __restrict__ XW, const float* __restrict__ a,
                              float* __restrict__ s_col, float* __restrict__ s_row) {
  __shared__ float red0[128];
  __shared__ float red1[128];
  int r = blockIdx.x, t = threadIdx.x;
  float x = XW[(size_t)r * NF + t];
  red0[t] = x * a[t];
  red1[t] = x * a[NF + t];
  __syncthreads();
#pragma unroll
  for (int s = 64; s > 0; s >>= 1) {
    if (t < s) { red0[t] += red0[t + s]; red1[t] += red1[t + s]; }
    __syncthreads();
  }
  if (t == 0) { s_col[r] = red0[0]; s_row[r] = red1[0]; }
}

// denT[j][k] = exp(leaky_relu(s_row[k] + s_col[j]))  (den TRANSPOSED, bf16)
__global__ void ga_denT(const float* __restrict__ s_row, const float* __restrict__ s_col,
                        __bf16* __restrict__ denT) {
  int k = blockIdx.x * blockDim.x + threadIdx.x;
  int j = blockIdx.y;
  float z = s_row[k] + s_col[j];
  z = z > 0.0f ? z : 0.01f * z;
  denT[(size_t)j * NN + k] = f2bf(__expf(z));
}

// ---------- WMMA fragment loaders (contiguous k = 16*(lane>>4) + s, wave32) ----------
// A operand 16x32 bf16 from LDS tile (row stride TS bf16)
static __device__ __forceinline__ v16bf ga_frag_a(const __bf16* tile) {
  int lane = threadIdx.x & 31;
  int row = lane & 15, hi = lane >> 4;
  union { v16bf v; unsigned int w[8]; } f;
  const __bf16* base = tile + row * TS;
#pragma unroll
  for (int v = 0; v < 4; ++v)
    f.w[v] = *reinterpret_cast<const unsigned int*>(base + 2 * v + hi * 16);
#pragma unroll
  for (int v = 4; v < 8; ++v)
    f.w[v] = *reinterpret_cast<const unsigned int*>(base + 8 + 2 * (v - 4) + hi * 16);
  return f.v;
}

// B operand 32x16 bf16 from LDS tile stored N-major (Bt[n][k], stride TS):
//   lanes 0-15: N=lane, VGPR v -> K=2v,2v+1 ; lanes 16-31: K += 16.
static __device__ __forceinline__ v16bf ga_frag_b(const __bf16* tileT, int ncol) {
  int lane = threadIdx.x & 31;
  int n = lane & 15, kh = lane >> 4;
  union { v16bf v; unsigned int w[8]; } f;
  const __bf16* base = tileT + (ncol + n) * TS + kh * 16;
#pragma unroll
  for (int v = 0; v < 8; ++v)
    f.w[v] = *reinterpret_cast<const unsigned int*>(base + 2 * v);
  return f.v;
}

// issue async copies for one 128x32 bf16 tile: 512 x 16B chunks, 2 per thread
// g points at element (rowbase, k0) of a row-major [rows x ld] bf16 matrix.
static __device__ __forceinline__ void ga_async_tile(const __bf16* __restrict__ g,
                                                     size_t ld, unsigned ldsbase, int tid) {
#pragma unroll
  for (int i = 0; i < 2; ++i) {
    int c = i * 256 + tid;
    int row = c >> 2, part = c & 3;
    unsigned long long ga =
        (unsigned long long)(size_t)(g + (size_t)row * ld + part * 8);
    unsigned la = ldsbase + (unsigned)(row * (TS * 2) + part * 16);
    asm volatile("global_load_async_to_lds_b128 %0, %1, off"
                 :: "v"(la), "v"(ga) : "memory");
  }
}

// ---------- 128x128-tile bf16 GEMM, fp32 accumulate, double-buffered async LDS ----
// A  : [Mdim x Kdim] row-major bf16
// BgT: [Ndim x Kdim] row-major bf16 (i.e. B transposed)
// mode 0: Cout (fp32, optional) and CbfT (bf16 transposed mirror [Ndim x Mdim], optional)
// mode 1: fused attention epilogue: att = den(i,j)/num, num = (A@B)(i,j)
__global__ __launch_bounds__(256) void ga_gemm_bf16(
    const __bf16* __restrict__ Ag, const __bf16* __restrict__ BgT,
    float* __restrict__ Cout, __bf16* __restrict__ CbfT,
    int Kdim, int Ndim, int Mdim, int mode,
    const float* __restrict__ s_row, const float* __restrict__ s_col,
    __bf16* __restrict__ attOut)
{
  __shared__ __bf16 As[2][128 * TS];
  __shared__ __bf16 Bs[2][128 * TS];

  const int tid  = threadIdx.x;
  const int wave = tid >> 5;
  const int lane = tid & 31;
  const int Mbase = blockIdx.y * 128;
  const int Nbase = blockIdx.x * 128;
  const int mt0 = (wave & 3) * 2;    // this wave: 2 M-tiles
  const int nt0 = (wave >> 2) * 4;   // this wave: 4 N-tiles

  const unsigned asL[2] = { (unsigned)(size_t)&As[0][0], (unsigned)(size_t)&As[1][0] };
  const unsigned bsL[2] = { (unsigned)(size_t)&Bs[0][0], (unsigned)(size_t)&Bs[1][0] };

  v8f zero = {};
  v8f acc[2][4];
#pragma unroll
  for (int mi = 0; mi < 2; ++mi)
#pragma unroll
    for (int ni = 0; ni < 4; ++ni) acc[mi][ni] = zero;

  // prologue: stage k-step 0 into buffer 0
  ga_async_tile(Ag  + (size_t)Mbase * Kdim, Kdim, asL[0], tid);
  ga_async_tile(BgT + (size_t)Nbase * Kdim, Kdim, bsL[0], tid);

  const int nsteps = Kdim >> 5;
  for (int s = 0; s < nsteps; ++s) {
    const int buf = s & 1;
    if (s + 1 < nsteps) {
      const int k1 = (s + 1) * 32;
      ga_async_tile(Ag  + (size_t)Mbase * Kdim + k1, Kdim, asL[buf ^ 1], tid);
      ga_async_tile(BgT + (size_t)Nbase * Kdim + k1, Kdim, bsL[buf ^ 1], tid);
      // 4 newest async ops may remain outstanding; older batch (this step) done
      asm volatile("s_wait_asynccnt 0x4" ::: "memory");
    } else {
      asm volatile("s_wait_asynccnt 0x0" ::: "memory");
    }
    __syncthreads();

    v16bf af0 = ga_frag_a(&As[buf][(mt0 * 16) * TS]);
    v16bf af1 = ga_frag_a(&As[buf][((mt0 + 1) * 16) * TS]);
#pragma unroll
    for (int ni = 0; ni < 4; ++ni) {
      v16bf bf = ga_frag_b(&Bs[buf][0], (nt0 + ni) * 16);
      acc[0][ni] = __builtin_amdgcn_wmma_f32_16x16x32_bf16(
          false, af0, false, bf, (short)0, acc[0][ni], false, false);
      acc[1][ni] = __builtin_amdgcn_wmma_f32_16x16x32_bf16(
          false, af1, false, bf, (short)0, acc[1][ni], false, false);
    }
    __syncthreads();   // all waves done reading buf before it is overwritten
  }

  // epilogue: C 16x16 f32 layout -> VGPR r: M = r + 8*(lane>>4), N = lane&15
  const int nl = lane & 15, mh = lane >> 4;
#pragma unroll
  for (int mi = 0; mi < 2; ++mi) {
#pragma unroll
    for (int ni = 0; ni < 4; ++ni) {
#pragma unroll
      for (int r = 0; r < 8; ++r) {
        int gm = Mbase + (mt0 + mi) * 16 + mh * 8 + r;
        int gn = Nbase + (nt0 + ni) * 16 + nl;
        float c = acc[mi][ni][r];
        if (mode == 0) {
          if (Cout) Cout[(size_t)gm * Ndim + gn] = c;
          if (CbfT) CbfT[(size_t)gn * Mdim + gm] = f2bf(c);
        } else {
          float z = s_row[gm] + s_col[gn];
          z = z > 0.0f ? z : 0.01f * z;
          float d = __expf(z);                    // den(i,j) recomputed in fp32
          float att = (c != 0.0f) ? d / c : 0.0f; // where(num!=0, den/num, 0)
          attOut[(size_t)gm * Ndim + gn] = f2bf(att);
        }
      }
    }
  }
}

// ---------- host launcher ----------
extern "C" void kernel_launch(void* const* d_in, const int* in_sizes, int n_in,
                              void* d_out, int out_size, void* d_ws, size_t ws_size,
                              hipStream_t stream) {
  const float* X = (const float*)d_in[0];   // [4096,128]
  const float* A = (const float*)d_in[1];   // [4096,4096]
  const float* W = (const float*)d_in[2];   // [128,128]
  const float* a = (const float*)d_in[3];   // [1,256]
  float* H = (float*)d_out;                 // [4096,128]

  char* ws = (char*)d_ws;
  size_t off = 0;
  auto alloc = [&](size_t bytes) -> void* {
    void* p = ws + off;
    off = (off + bytes + 255) & ~(size_t)255;
    return p;
  };

  __bf16* Xbf   = (__bf16*)alloc((size_t)NN * NF * 2);
  __bf16* WbfT  = (__bf16*)alloc((size_t)NF * NF * 2);   // W transposed [NF x NF]
  __bf16* Abf   = (__bf16*)alloc((size_t)NN * NN * 2);
  float*  XW    = (float*) alloc((size_t)NN * NF * 4);
  __bf16* XWbfT = (__bf16*)alloc((size_t)NF * NN * 2);   // XW transposed [128 x 4096]
  float*  s_col = (float*) alloc((size_t)NN * 4);
  float*  s_row = (float*) alloc((size_t)NN * 4);
  __bf16* denT  = (__bf16*)alloc((size_t)NN * NN * 2);   // den transposed
  __bf16* attbf = (__bf16*)alloc((size_t)NN * NN * 2);   // att row-major (A operand)
  __bf16* MbfT  = (__bf16*)alloc((size_t)NF * NN * 2);   // (att@XW) transposed
  (void)ws_size; (void)in_sizes; (void)n_in; (void)out_size;

  // 1) conversions (all B operands are produced pre-transposed)
  ga_cvt_bf16<<<(NN * NF + 255) / 256, 256, 0, stream>>>(X, Xbf, NN * NF);
  ga_cvt_bf16<<<NN * NN / 256, 256, 0, stream>>>(A, Abf, NN * NN);
  ga_cvtT_bf16<<<(NF * NF + 255) / 256, 256, 0, stream>>>(W, WbfT, NF, NF);

  // 2) XW = X @ W  (M=4096, N=128, K=128); also emit XW^T bf16 mirror
  ga_gemm_bf16<<<dim3(1, NN / 128), 256, 0, stream>>>(
      Xbf, WbfT, XW, XWbfT, NF, NF, NN, 0, nullptr, nullptr, nullptr);

  // 3) attention scores
  ga_row_scores<<<NN, 128, 0, stream>>>(XW, a, s_col, s_row);

  // 4) den^T = exp(leaky_relu(...))^T in bf16 (L2-resident B operand)
  ga_denT<<<dim3(NN / 256, NN), 256, 0, stream>>>(s_row, s_col, denT);

  // 5) num = A @ den with fused att = den/num epilogue (num never hits memory)
  ga_gemm_bf16<<<dim3(NN / 128, NN / 128), 256, 0, stream>>>(
      Abf, denT, nullptr, nullptr, NN, NN, NN, 1, s_row, s_col, attbf);

  // 6) M = att @ XW  (M=4096, N=128, K=4096); only the M^T bf16 mirror is needed
  ga_gemm_bf16<<<dim3(1, NN / 128), 256, 0, stream>>>(
      attbf, XWbfT, nullptr, MbfT, NN, NF, NN, 0, nullptr, nullptr, nullptr);

  // 7) H = A @ M     (M=4096, N=128, K=4096) -> fp32 output
  ga_gemm_bf16<<<dim3(1, NN / 128), 256, 0, stream>>>(
      Abf, MbfT, H, nullptr, NN, NF, NN, 0, nullptr, nullptr, nullptr);
}
